// Self_Attn_20358144983355
// MI455X (gfx1250) — matrix operand order repaired and checked by reading, not compiled, not run on hardware
//
#include <hip/hip_runtime.h>

typedef _Float16 half_t;
typedef _Float16 v16h __attribute__((ext_vector_type(16)));
typedef float    v8f  __attribute__((ext_vector_type(8)));

// ---------------- problem sizes ----------------
// x: (8,64,70,70) f32 ; convs 7x7 VALID -> 64x64 ; N=4096, M=1024 after 2x2 pool
#define BATCH   8
#define CIN     64
#define HIN     70
#define WIN     70
#define HOUT    64
#define NPOS    4096      // 64*64
#define MPOS    1024      // 32*32
#define KPAD    4096      // 64 channels * (7x7 padded to 8x8) -> pure shift/mask decode
#define XELEMS  (BATCH*CIN*HIN*WIN)          // 2,508,800
#define XPB     (CIN*HIN*WIN)                // 313,600 per batch
#define CTH     100
#define CG      200
#define CTH_PAD 128       // K pad for attn GEMM (multiple of 32)
#define CG_PAD  208       // row pad for gemm2 (multiple of 16)

// conv LDS slab: x[c][7 rows][70 cols] halves + overrun pad for the 8x8 filter pad reads
#define SLAB_ELEMS (CIN*7*WIN + 96)          // 31,360 + 96 -> 62,912 B

// ---------------- workspace layout (bytes) ----------------
#define OFF_XH    ((size_t)0)                               // f16 x            5,017,600
#define OFF_WTH   (OFF_XH   + (size_t)XELEMS*2)             // f16 theta_w pad    819,200
#define OFF_WPH   (OFF_WTH  + (size_t)CTH*KPAD*2)           // f16 phi_w pad      819,200
#define OFF_WG    (OFF_WPH  + (size_t)CTH*KPAD*2)           // f16 g_w pad      1,638,400
#define OFF_TRAW  (OFF_WG   + (size_t)CG*KPAD*2)            // f16 [b][100][4096]
#define OFF_PRAW  (OFF_TRAW + (size_t)BATCH*CTH*NPOS*2)
#define OFF_GRAW  (OFF_PRAW + (size_t)BATCH*CTH*NPOS*2)
#define OFF_TND   (OFF_GRAW + (size_t)BATCH*CG*NPOS*2)      // f16 [b][4096][128]
#define OFF_PP    (OFF_TND  + (size_t)BATCH*NPOS*CTH_PAD*2) // f16 [b][1024][128] (transposed!)
#define OFF_GP    (OFF_PP   + (size_t)BATCH*MPOS*CTH_PAD*2) // f16 [b][208][1024]
#define OFF_ATT   (OFF_GP   + (size_t)BATCH*CG_PAD*MPOS*2)  // f16 [b][4096][1024]

// ---------------- WMMA helpers (CDNA5 16x16x32 f16 -> f32) ----------------
__device__ __forceinline__ v8f wmma16(v16h a, v16h b, v8f c) {
    // (neg_a, A, neg_b, B, c_mod, C, reuse_a, reuse_b)
    return __builtin_amdgcn_wmma_f32_16x16x32_f16(false, a, false, b, (short)0, c, false, false);
}

// A fragment (16x32, 16-bit): element t of lane L (hi = L>=16) holds
// K = hi*16 + t ; pairs (2v,2v+1) contiguous in K -> b128-mergeable
__device__ __forceinline__ v16h load_afrag(const half_t* __restrict__ rowK, int hi) {
    v16h a;
#pragma unroll
    for (int v = 0; v < 8; ++v) {
        int kk = (hi << 4) + (v << 1);
        a[2*v]   = rowK[kk];
        a[2*v+1] = rowK[kk+1];
    }
    return a;
}

// B fragment (32x16, 16-bit), K contiguous per half-wave: element t -> K = hi*16 + t
__device__ __forceinline__ v16h load_bfrag_contig(const half_t* __restrict__ colK, int hi) {
    v16h b;
#pragma unroll
    for (int t = 0; t < 16; ++t) b[t] = colK[(hi << 4) + t];
    return b;
}

// ---------------- utility kernels ----------------
__global__ void f32_to_f16(const float* __restrict__ src, half_t* __restrict__ dst, int n) {
    int i = blockIdx.x * blockDim.x + threadIdx.x;
    if (i < n) dst[i] = (half_t)src[i];
}

// pack conv weights [outC][64][7][7] f32 -> [outC][4096] f16 with each 7x7 padded to 8x8
__global__ void pack_w(const float* __restrict__ src, half_t* __restrict__ dst, int outC) {
    int i = blockIdx.x * blockDim.x + threadIdx.x;
    if (i >= outC * KPAD) return;
    int oc = i >> 12, k = i & (KPAD - 1);
    int c = k >> 6, kh = (k >> 3) & 7, kw = k & 7;
    half_t v = (half_t)0.f;
    if (kh < 7 && kw < 7) v = (half_t)src[((oc * CIN + c) * 7 + kh) * 7 + kw];
    dst[i] = v;
}

// ---------------- conv 7x7 VALID as implicit-im2col WMMA GEMM ----------------
// grid: (64 oh rows, ceil(outC/16), BATCH), block 128 (4 waves; wave -> 16oc x 16ow tile)
// Input slab for the whole workgroup staged in LDS once; K padded to 4096 (shift/mask decode).
__global__ void conv7_wmma(const half_t* __restrict__ xh, const half_t* __restrict__ wh,
                           const float* __restrict__ bias, half_t* __restrict__ out, int outC) {
    __shared__ half_t slab[SLAB_ELEMS];                    // 62,912 B
    const int tid = threadIdx.x;
    const int b = blockIdx.z, oc0 = blockIdx.y * 16;
    const int oh = blockIdx.x;

    // stage x[b][c][oh+0..6][0..69] -> slab[c*490 + kh*70 + col]  (dword copies)
    {
        const unsigned int* xsrc = (const unsigned int*)(xh + (size_t)b * XPB);
        unsigned int* sl = (unsigned int*)slab;
        for (int i = tid; i < 448 * 35; i += 128) {        // 448 rows x 35 dwords
            int r = i / 35, cd = i - r * 35;
            int c = r / 7,  kh = r - c * 7;
            sl[r * 35 + cd] = xsrc[c * 2450 + (oh + kh) * 35 + cd];
        }
    }
    __syncthreads();

    const int lane = tid & 31, wid = tid >> 5;
    const int hi = lane >> 4, col = lane & 15;
    const int ow = wid * 16 + col;
    // A-matrix row = lane%16; clamp out-of-range oc rows to row 0 (always in-bounds);
    // their garbage accumulators are discarded by the store guard below.
    const int arow = (oc0 + col < outC) ? (oc0 + col) : 0;
    const half_t* wrow = wh + (size_t)arow * KPAD;

    v8f acc = {};
    for (int k0 = 0; k0 < KPAD; k0 += 32) {
        v16h a = load_afrag(wrow + k0, hi);
        // K-base per half-wave is 16-aligned -> c and kh-base are uniform over t
        int kbase = k0 + (hi << 4);
        int base = (kbase >> 6) * 490 + ((kbase >> 3) & 7) * 70 + ow;
        v16h bf;
#pragma unroll
        for (int t = 0; t < 16; ++t)
            bf[t] = slab[base + ((t >> 3) * 70) + (t & 7)];
        acc = wmma16(a, bf, acc);
    }
#pragma unroll
    for (int r = 0; r < 8; ++r) {
        int oc = oc0 + r + (hi << 3);
        if (oc < outC)
            out[((size_t)b * outC + oc) * NPOS + oh * HOUT + ow] = (half_t)(acc[r] + bias[oc]);
    }
}

// ---------------- packing: theta transpose+pad, phi/g maxpool+pad ----------------
__global__ void pack_theta(const half_t* __restrict__ raw, half_t* __restrict__ dst) {
    int i = blockIdx.x * blockDim.x + threadIdx.x;         // [b][n][c], c in 0..127
    if (i >= BATCH * NPOS * CTH_PAD) return;
    int c = i & (CTH_PAD - 1);
    int bn = i / CTH_PAD;
    int n = bn & (NPOS - 1);
    int b = bn / NPOS;
    dst[i] = (c < CTH) ? raw[((size_t)b * CTH + c) * NPOS + n] : (half_t)0.f;
}

__device__ __forceinline__ half_t pool4(const half_t* __restrict__ base) {
    float v0 = (float)base[0], v1 = (float)base[1];
    float v2 = (float)base[HOUT], v3 = (float)base[HOUT + 1];
    return (half_t)fmaxf(fmaxf(v0, v1), fmaxf(v2, v3));
}

// phi: maxpool + transpose -> PP[b][m][c] so gemm1 B-fragments are K-contiguous
__global__ void pack_phi_t(const half_t* __restrict__ raw, half_t* __restrict__ dst) {
    int i = blockIdx.x * blockDim.x + threadIdx.x;         // [b][m][c]
    if (i >= BATCH * MPOS * CTH_PAD) return;
    int c = i & (CTH_PAD - 1);
    int bm = i / CTH_PAD;
    int m = bm & (MPOS - 1);
    int b = bm / MPOS;
    half_t v = (half_t)0.f;
    if (c < CTH) {
        int ph = m >> 5, pw = m & 31;
        v = pool4(raw + ((size_t)b * CTH + c) * NPOS + (ph * 2) * HOUT + pw * 2);
    }
    dst[i] = v;
}

// g: maxpool -> GP[b][ct][m] (row-major ct; padded rows zero)
__global__ void pack_g(const half_t* __restrict__ raw, half_t* __restrict__ dst) {
    int i = blockIdx.x * blockDim.x + threadIdx.x;         // [b][ct][m]
    if (i >= BATCH * CG_PAD * MPOS) return;
    int m = i & (MPOS - 1);
    int bc = i / MPOS;
    int c = bc % CG_PAD;
    int b = bc / CG_PAD;
    half_t v = (half_t)0.f;
    if (c < CG) {
        int ph = m >> 5, pw = m & 31;
        v = pool4(raw + ((size_t)b * CG + c) * NPOS + (ph * 2) * HOUT + pw * 2);
    }
    dst[i] = v;
}

// ---------------- GEMM1 (theta^T x phi) fused with row softmax ----------------
// grid: (NPOS/16, 1, BATCH), block 128; LDS holds full 16x1024 f32 score stripe
__global__ void gemm1_softmax(const half_t* __restrict__ tnd, const half_t* __restrict__ pp,
                              half_t* __restrict__ att) {
    __shared__ float scores[16 * MPOS];                    // 64 KB
    const int lane = threadIdx.x & 31, wid = threadIdx.x >> 5;
    const int hi = lane >> 4, col = lane & 15;
    const int b = blockIdx.z, n0 = blockIdx.x * 16;

    // A fragments depend only on K -> hoist all 4 (K = c over padded 128 channels)
    const half_t* arow = tnd + ((size_t)b * NPOS + n0 + col) * CTH_PAD;
    v16h afr[4];
#pragma unroll
    for (int kc = 0; kc < 4; ++kc) afr[kc] = load_afrag(arow + kc * 32, hi);

    const half_t* pb = pp + (size_t)b * MPOS * CTH_PAD;    // [m][c]
    for (int mt = 0; mt < 16; ++mt) {
        int m0 = wid * 256 + mt * 16;
        const half_t* brow = pb + (size_t)(m0 + col) * CTH_PAD;
        __builtin_prefetch(brow + 16 * CTH_PAD, 0, 1);     // next m-tile's row
        v8f acc = {};
#pragma unroll
        for (int kc = 0; kc < 4; ++kc)
            acc = wmma16(afr[kc], load_bfrag_contig(brow + kc * 32, hi), acc);
#pragma unroll
        for (int r = 0; r < 8; ++r)
            scores[(r + (hi << 3)) * MPOS + m0 + col] = acc[r];
    }
    __syncthreads();

    // each wave softmaxes 4 rows with wave32 butterfly reductions
    for (int rr = 0; rr < 4; ++rr) {
        int row = wid * 4 + rr;
        float* srow = scores + row * MPOS;
        float mx = -3.4e38f;
        for (int c = lane; c < MPOS; c += 32) mx = fmaxf(mx, srow[c]);
#pragma unroll
        for (int off = 16; off > 0; off >>= 1) mx = fmaxf(mx, __shfl_xor(mx, off, 32));
        float sum = 0.f;
        for (int c = lane; c < MPOS; c += 32) {
            float e = __expf(srow[c] - mx);
            srow[c] = e;
            sum += e;
        }
#pragma unroll
        for (int off = 16; off > 0; off >>= 1) sum += __shfl_xor(sum, off, 32);
        float inv = 1.f / sum;
        half_t* orow = att + ((size_t)b * NPOS + n0 + row) * MPOS;
        for (int c = lane; c < MPOS; c += 32) orow[c] = (half_t)(srow[c] * inv);
    }
}

// ---------------- GEMM2: out[b][ct][n] = sum_m g[b][ct][m] * attn[b][n][m] ----------------
// grid: (NPOS/64, CG_PAD/16, BATCH), block 128 (4 waves, one 16x16 tile each), K = 1024
__global__ void gemm2_wmma(const half_t* __restrict__ gp, const half_t* __restrict__ att,
                           float* __restrict__ out) {
    const int lane = threadIdx.x & 31, wid = threadIdx.x >> 5;
    const int hi = lane >> 4, col = lane & 15;
    const int b = blockIdx.z, ct0 = blockIdx.y * 16;
    const int n0 = blockIdx.x * 64 + wid * 16;
    const half_t* arow = gp  + ((size_t)b * CG_PAD + ct0 + col) * MPOS;  // padded rows are 0
    const half_t* brow = att + ((size_t)b * NPOS   + n0  + col) * MPOS;  // K = m contiguous
    v8f acc = {};
    for (int k0 = 0; k0 < MPOS; k0 += 32) {
        __builtin_prefetch(arow + k0 + 256, 0, 1);
        __builtin_prefetch(brow + k0 + 256, 0, 1);
        v16h a  = load_afrag(arow + k0, hi);
        v16h bf = load_bfrag_contig(brow + k0, hi);
        acc = wmma16(a, bf, acc);
    }
#pragma unroll
    for (int r = 0; r < 8; ++r) {
        int ct = ct0 + r + (hi << 3);
        if (ct < CG)
            out[((size_t)b * CG + ct) * NPOS + n0 + col] = acc[r];
    }
}

// ---------------- launcher ----------------
extern "C" void kernel_launch(void* const* d_in, const int* in_sizes, int n_in,
                              void* d_out, int out_size, void* d_ws, size_t ws_size,
                              hipStream_t stream) {
    const float* x  = (const float*)d_in[0];
    const float* tw = (const float*)d_in[1];
    const float* tb = (const float*)d_in[2];
    const float* pw = (const float*)d_in[3];
    const float* pb = (const float*)d_in[4];
    const float* gw = (const float*)d_in[5];
    const float* gb = (const float*)d_in[6];
    float* out = (float*)d_out;
    char* ws = (char*)d_ws;

    half_t* XH   = (half_t*)(ws + OFF_XH);
    half_t* WTH  = (half_t*)(ws + OFF_WTH);
    half_t* WPH  = (half_t*)(ws + OFF_WPH);
    half_t* WG   = (half_t*)(ws + OFF_WG);
    half_t* TRAW = (half_t*)(ws + OFF_TRAW);
    half_t* PRAW = (half_t*)(ws + OFF_PRAW);
    half_t* GRAW = (half_t*)(ws + OFF_GRAW);
    half_t* TND  = (half_t*)(ws + OFF_TND);
    half_t* PP   = (half_t*)(ws + OFF_PP);
    half_t* GP   = (half_t*)(ws + OFF_GP);
    half_t* ATT  = (half_t*)(ws + OFF_ATT);

    // precision conversion + weight padding (7x7 -> 8x8, K = 4096)
    f32_to_f16<<<(XELEMS + 255) / 256, 256, 0, stream>>>(x, XH, XELEMS);
    pack_w<<<(CTH * KPAD + 255) / 256, 256, 0, stream>>>(tw, WTH, CTH);
    pack_w<<<(CTH * KPAD + 255) / 256, 256, 0, stream>>>(pw, WPH, CTH);
    pack_w<<<(CG  * KPAD + 255) / 256, 256, 0, stream>>>(gw, WG,  CG);

    dim3 blk(128);
    // three convs as LDS-staged implicit-im2col WMMA GEMMs (K = 4096 = 128 chunks of 32)
    conv7_wmma<<<dim3(HOUT, 7, BATCH),  blk, 0, stream>>>(XH, WTH, tb, TRAW, CTH);
    conv7_wmma<<<dim3(HOUT, 7, BATCH),  blk, 0, stream>>>(XH, WPH, pb, PRAW, CTH);
    conv7_wmma<<<dim3(HOUT, 13, BATCH), blk, 0, stream>>>(XH, WG,  gb, GRAW, CG);

    // repack for the attention GEMMs (transpose / 2x2 maxpool / zero-pad)
    pack_theta<<<(BATCH * NPOS * CTH_PAD + 255) / 256, 256, 0, stream>>>(TRAW, TND);
    pack_phi_t<<<(BATCH * MPOS * CTH_PAD + 255) / 256, 256, 0, stream>>>(PRAW, PP);
    pack_g<<<(BATCH * CG_PAD * MPOS + 255) / 256, 256, 0, stream>>>(GRAW, GP);

    // attn = softmax(theta^T phi) fused in LDS ; then out = g x attn^T
    gemm1_softmax<<<dim3(NPOS / 16, 1, BATCH), blk, 0, stream>>>(TND, PP, ATT);
    gemm2_wmma<<<dim3(NPOS / 64, CG_PAD / 16, BATCH), blk, 0, stream>>>(GP, ATT, out);
}
